// LaterallyConnectedLayer_24816321036622
// MI455X (gfx1250) — matrix operand order repaired and checked by reading, not compiled, not run on hardware
//
#include <hip/hip_runtime.h>
#include <hip/hip_bf16.h>

// ---------------------------------------------------------------------------
// LaterallyConnectedLayer for MI455X (gfx1250, wave32, WMMA).
//
//  1. conv1 + minmax + softmax eliminated: softmax over each fm sums to 1,
//     so s[b,c] = sum(Ab[b,c]) + 1 and winners = argmax_n sum(Ab).
//  2. conv2 compacted to winner channels (per-batch GEMM M=32,K=800,N=3136).
//  3. einsum + conv2 on v_wmma_f32_16x16x32_f16, branch-free inner loops:
//     - einsum: zero-haloed 64x64 images, channel-major, so the (xo,yo)
//       window shift is a constant offset on the flat K index; clamped base,
//       halo zeros on the A operand kill all out-of-window products.
//       K split 4-way across blocks (partials + deterministic reduce) for
//       4x the wave count on the dominant kernel.
//     - conv2: K ordered kp*32+fi with activations in [b][60][60][32] so both
//       fragments are aligned contiguous (b128-mergeable).
//  Workspace ~75.5 MB (fits in 192 MB L2).
// ---------------------------------------------------------------------------

typedef __attribute__((ext_vector_type(16))) _Float16 v16h;
typedef __attribute__((ext_vector_type(8)))  float    v8f;

#define B_   16
#define NFM  32
#define C_   128
#define H_   56
#define W_   56
#define HW   (H_*W_)          // 3136
#define KSZ  5
#define KK   (KSZ*KSZ)        // 25
#define PADW 60               // conv pad width
#define PIMG 64               // haloed image width (einsum)
#define PPIX (PIMG*PIMG)      // 4096
#define CHS  (B_*PPIX)        // 65536: per-channel stride in Pab/Amp
#define NCHUNK 4              // einsum K-split factor
#define KCN  (C_*C_*KK)       // 409600: K_change element count
#define EPS_ 1e-10f

// symm_pad(x, 2) composite index map (56 -> 56)
__device__ __forceinline__ int map2(int i) {
  if (i < 2)   return 3 - i;
  if (i >= 54) return 107 - i;
  return i;
}
// pad_activations composite map: padded coord (0..59) -> Ab coord (0..55)
__device__ __forceinline__ int mapP(int r) {
  if (r < 4)  return 5 - r;
  if (r > 55) return 109 - r;
  return r - 2;
}

static __device__ __forceinline__ v8f wmma_f16(v16h a, v16h b, v8f c) {
  return __builtin_amdgcn_wmma_f32_16x16x32_f16(false, a, false, b,
                                                (short)0, c, false, false);
}

// --------------------------------------------------------------------------
// Ab = clip(tile(symm_pad(A,2), 4) + 0.1*noise, 0)  (f32, [b][c][56][56])
// --------------------------------------------------------------------------
__global__ void __launch_bounds__(256) build_ab(
    const float* __restrict__ A, const float* __restrict__ noise,
    float* __restrict__ Ab)
{
  int idx = blockIdx.x * blockDim.x + threadIdx.x;
  if (idx >= B_ * C_ * HW) return;
  int hw = idx % HW;
  int c  = (idx / HW) % C_;
  int b  = idx / (HW * C_);
  int h = hw / W_, w = hw % W_;
  float v = A[((b * NFM + (c & 31)) * H_ + map2(h)) * W_ + map2(w)]
          + 0.1f * noise[idx];
  Ab[idx] = fmaxf(v, 0.0f);
}

// --------------------------------------------------------------------------
// Pab[c][b][64][64] (f16): Ab at offset (4,4), zero halo elsewhere.
// --------------------------------------------------------------------------
__global__ void __launch_bounds__(256) build_pab(
    const float* __restrict__ Ab, _Float16* __restrict__ Pab)
{
  int idx = blockIdx.x * blockDim.x + threadIdx.x;
  if (idx >= C_ * CHS) return;
  int pos = idx & (PPIX - 1);
  int b   = (idx >> 12) & (B_ - 1);
  int c   = idx >> 16;
  int h = (pos >> 6) - 4, w = (pos & 63) - 4;
  float v = 0.0f;
  if ((unsigned)h < (unsigned)H_ && (unsigned)w < (unsigned)W_)
    v = Ab[((b * C_ + c) * H_ + h) * W_ + w];
  Pab[idx] = (_Float16)v;
}

// --------------------------------------------------------------------------
// s[b,c] = sum_{h,w} Ab   (one block per fm; fixed-order tree -> determinism)
// --------------------------------------------------------------------------
__global__ void __launch_bounds__(256) rowsum_kernel(
    const float* __restrict__ Ab, float* __restrict__ s)
{
  __shared__ float sm[256];
  int bc = blockIdx.x;
  const float* p = Ab + (size_t)bc * HW;
  float acc = 0.0f;
  for (int i = threadIdx.x; i < HW; i += 256) acc += p[i];
  sm[threadIdx.x] = acc;
  __syncthreads();
  for (int st = 128; st > 0; st >>= 1) {
    if (threadIdx.x < st) sm[threadIdx.x] += sm[threadIdx.x + st];
    __syncthreads();
  }
  if (threadIdx.x == 0) s[bc] = sm[0];
}

// --------------------------------------------------------------------------
// winners[b,fm] = argmax_n s[b, n*32+fm] (first-max tie-break like jnp)
// --------------------------------------------------------------------------
__global__ void __launch_bounds__(256) argmax_kernel(
    const float* __restrict__ s, int* __restrict__ winners,
    int* __restrict__ fmidx)
{
  int t = blockIdx.x * blockDim.x + threadIdx.x;
  if (t >= B_ * NFM) return;
  int b = t / NFM, fm = t % NFM;
  int best = 0;
  float bv = s[b * C_ + fm];
  for (int n = 1; n < 4; ++n) {
    float v = s[b * C_ + n * NFM + fm];
    if (v > bv) { bv = v; best = n; }
  }
  winners[t] = best;
  fmidx[t]   = best * NFM + fm;
}

// --------------------------------------------------------------------------
// Amp = Pab * win_mask  (same [c][b][64][64] layout)
// --------------------------------------------------------------------------
__global__ void __launch_bounds__(256) build_amp(
    const _Float16* __restrict__ Pab, const int* __restrict__ winners,
    _Float16* __restrict__ Amp)
{
  int idx = blockIdx.x * blockDim.x + threadIdx.x;
  if (idx >= C_ * CHS) return;
  int b = (idx >> 12) & (B_ - 1);
  int c = idx >> 16;
  bool m = (winners[b * NFM + (c & 31)] == (c >> 5));
  Amp[idx] = m ? Pab[idx] : (_Float16)0.0f;
}

// --------------------------------------------------------------------------
// Partial K_change: KcP[z][c,a,off] = sum_{k in chunk z} Amp[c,k]*Pab[a,k-shift]
// shift = xo*64+yo. Halo zeros in Amp kill out-of-window/cross-image terms;
// base clamp keeps all reads inside the finite Pab buffer.
// One wave per 16(M) x 32(N) tile per offset per K-chunk; 2 wmma/step.
// Fragment k order: contiguous k = 16*(lane/16) + s, same for A and B (wave32).
// --------------------------------------------------------------------------
__global__ void __launch_bounds__(32) einsum_wmma(
    const _Float16* __restrict__ Amp, const _Float16* __restrict__ Pab,
    float* __restrict__ KcP)
{
  const int lane = threadIdx.x;
  const int mt  = blockIdx.x >> 2;               // 0..7   (c tile)
  const int ntp = blockIdx.x & 3;                // 0..3   (a tile pair)
  const int off = blockIdx.y;                    // 0..24
  const int kbeg = blockIdx.z * (CHS / NCHUNK);  // K chunk
  const int kend = kbeg + (CHS / NCHUNK);
  const int xo = off / KSZ - 2, yo = off % KSZ - 2;
  const int shift = xo * PIMG + yo;              // [-130, 130]
  const int crow  = mt * 16 + (lane & 15);
  const int acol0 = ntp * 32 + (lane & 15);
  const int acol1 = acol0 + 16;
  const int aoff = (lane < 16) ? 0 : 16;
  const int boff = (lane < 16) ? 0 : 16;
  const _Float16* arow = Amp + crow * CHS;
  const int bbase0 = acol0 * CHS + boff - shift;
  const int bbase1 = acol1 * CHS + boff - shift;
  const int TOTC = C_ * CHS;
  v8f acc0 = {}, acc1 = {};
  for (int k0 = kbeg; k0 < kend; k0 += 32) {
    v16h va, vb0, vb1;
    const _Float16* ap = arow + k0 + aoff;       // 16B-aligned chunks
#pragma unroll
    for (int j = 0; j < 8; ++j) { va[j] = ap[j]; va[j + 8] = ap[j + 8]; }
    int i0 = bbase0 + k0;
    int i1 = bbase1 + k0;
    i0 = (i0 < 0) ? 0 : ((i0 > TOTC - 16) ? TOTC - 16 : i0);   // branchless
    i1 = (i1 < 0) ? 0 : ((i1 > TOTC - 16) ? TOTC - 16 : i1);
    const _Float16* bp0 = Pab + i0;
    const _Float16* bp1 = Pab + i1;
#pragma unroll
    for (int j = 0; j < 16; ++j) { vb0[j] = bp0[j]; vb1[j] = bp1[j]; }
    acc0 = wmma_f16(va, vb0, acc0);
    acc1 = wmma_f16(va, vb1, acc1);
  }
  float* out = KcP + blockIdx.z * KCN;
#pragma unroll
  for (int r = 0; r < 8; ++r) {
    int c = mt * 16 + r + ((lane < 16) ? 0 : 8);
    out[(c * C_ + acol0) * KK + off] = acc0[r];
    out[(c * C_ + acol1) * KK + off] = acc1[r];
  }
}

// --------------------------------------------------------------------------
// Kc = (KcP[0]+KcP[1]+KcP[2]+KcP[3]) / 2048   (fixed order -> deterministic)
// --------------------------------------------------------------------------
__global__ void __launch_bounds__(256) reduce_kc(
    const float* __restrict__ KcP, float* __restrict__ Kc)
{
  int idx = blockIdx.x * blockDim.x + threadIdx.x;
  if (idx >= KCN) return;
  float v = KcP[idx];
  v += KcP[idx + KCN];
  v += KcP[idx + 2 * KCN];
  v += KcP[idx + 3 * KCN];
  Kc[idx] = v * (1.0f / 2048.0f);
}

// --------------------------------------------------------------------------
// Kn2[o,i,p] = minmax( 0.9*K[o,i,p] + 0.1*minmax(K_change[i,o,p]) )  (f16)
// --------------------------------------------------------------------------
__global__ void __launch_bounds__(256) make_kn2(
    const float* __restrict__ K, const float* __restrict__ Kc,
    _Float16* __restrict__ Kn2)
{
  int t = blockIdx.x * blockDim.x + threadIdx.x;
  if (t >= C_ * C_) return;
  int o = t / C_, i = t % C_;
  float kc[KK], t2[KK];
  float mn = 1e30f, mx = -1e30f;
  const float* pc = Kc + (i * C_ + o) * KK;
  for (int p = 0; p < KK; ++p) {
    kc[p] = pc[p];
    mn = fminf(mn, kc[p]); mx = fmaxf(mx, kc[p]);
  }
  float inv = 1.0f / (mx - mn + EPS_);
  const float* pk = K + (o * C_ + i) * KK;
  float mn2 = 1e30f, mx2 = -1e30f;
  for (int p = 0; p < KK; ++p) {
    t2[p] = 0.9f * pk[p] + 0.1f * ((kc[p] - mn) * inv);
    mn2 = fminf(mn2, t2[p]); mx2 = fmaxf(mx2, t2[p]);
  }
  float inv2 = 1.0f / (mx2 - mn2 + EPS_);
  _Float16* out = Kn2 + (o * C_ + i) * KK;
  for (int p = 0; p < KK; ++p) out[p] = (_Float16)((t2[p] - mn2) * inv2);
}

// --------------------------------------------------------------------------
// Wg[b][fo][kp*32+fi] = Kn2[fmidx[b,fo], fmidx[b,fi], kp]
// (K ordered kp-major so conv2's B fragment is fi-contiguous)
// --------------------------------------------------------------------------
__global__ void __launch_bounds__(256) gather_w(
    const _Float16* __restrict__ Kn2, const int* __restrict__ fmidx,
    _Float16* __restrict__ Wg)
{
  int t = blockIdx.x * blockDim.x + threadIdx.x;
  if (t >= B_ * NFM * NFM) return;
  int fi = t % NFM;
  int fo = (t / NFM) % NFM;
  int b  = t / (NFM * NFM);
  int och = fmidx[b * NFM + fo];
  int ich = fmidx[b * NFM + fi];
  const _Float16* src = Kn2 + (och * C_ + ich) * KK;
  _Float16* dst = Wg + (b * NFM + fo) * (NFM * KK) + fi;
  for (int p = 0; p < KK; ++p) dst[p * NFM] = src[p];
}

// --------------------------------------------------------------------------
// pA2t[b][r][c][fi] = pad_activations(A2)[b, fmidx[b,fi], r, c]  (f16)
// --------------------------------------------------------------------------
__global__ void __launch_bounds__(256) build_pa2t(
    const float* __restrict__ Ab, const int* __restrict__ fmidx,
    _Float16* __restrict__ pA2t)
{
  int idx = blockIdx.x * blockDim.x + threadIdx.x;
  if (idx >= B_ * PADW * PADW * NFM) return;
  int fi = idx & (NFM - 1);
  int rc = (idx >> 5) % (PADW * PADW);
  int b  = idx / (PADW * PADW * NFM);
  int r = rc / PADW, cc = rc % PADW;
  int ch = fmidx[b * NFM + fi];
  pA2t[idx] = (_Float16)Ab[((b * C_ + ch) * H_ + mapP(r)) * W_ + mapP(cc)];
}

// --------------------------------------------------------------------------
// Compacted conv2: L2c[b,fo,n] = (1/32)*sum_{kp,fi} Wg[b,fo,kp*32+fi] *
// pA2t[b, h+kh, w+kw, fi].  One wave per batch per 16-col tile; both M tiles
// in-wave (B fragment reused, 2 wmma per step); all loads aligned-contiguous.
// --------------------------------------------------------------------------
__global__ void __launch_bounds__(32) conv2_wmma(
    const _Float16* __restrict__ Wg, const _Float16* __restrict__ pA2t,
    float* __restrict__ L2)
{
  const int lane = threadIdx.x;
  const int nt = blockIdx.x;                 // 0..195
  const int b  = blockIdx.y;                 // 0..15
  const int n  = nt * 16 + (lane & 15);
  const int h  = n / W_, w = n - h * W_;
  const int aoff = (lane < 16) ? 0 : 16;
  const int boff = (lane < 16) ? 0 : 16;
  const _Float16* wrow0 = Wg + (b * NFM + (lane & 15)) * (NFM * KK);
  const _Float16* wrow1 = wrow0 + 16 * (NFM * KK);
  const _Float16* bbase = pA2t + (b * PADW * PADW + h * PADW + w) * NFM + boff;
  v8f acc0 = {}, acc1 = {};
  for (int kp = 0; kp < KK; ++kp) {
    int k0 = kp * NFM;
    int kh = kp / KSZ, kw = kp - kh * KSZ;
    const _Float16* ap0 = wrow0 + k0 + aoff;
    const _Float16* ap1 = wrow1 + k0 + aoff;
    const _Float16* bp  = bbase + (kh * PADW + kw) * NFM;  // 32B aligned
    v16h va0, va1, vb;
#pragma unroll
    for (int j = 0; j < 8; ++j) {
      va0[j] = ap0[j]; va0[j + 8] = ap0[j + 8];
      va1[j] = ap1[j]; va1[j + 8] = ap1[j + 8];
      vb[j]  = bp[j];  vb[j + 8]  = bp[j + 8];
    }
    acc0 = wmma_f16(va0, vb, acc0);
    acc1 = wmma_f16(va1, vb, acc1);
  }
#pragma unroll
  for (int r = 0; r < 8; ++r) {
    int fo0 = r + ((lane < 16) ? 0 : 8);
    L2[(b * NFM + fo0) * HW + n]      = acc0[r] * (1.0f / (float)NFM);
    L2[(b * NFM + fo0 + 16) * HW + n] = acc1[r] * (1.0f / (float)NFM);
  }
}

// --------------------------------------------------------------------------
// O[b,f,h,w] = Ab[b, fmidx[b,f], h, w] + 0.1 * L2c[b,f,h,w]
// --------------------------------------------------------------------------
__global__ void __launch_bounds__(256) final_out(
    const float* __restrict__ Ab, const float* __restrict__ L2,
    const int* __restrict__ fmidx, float* __restrict__ O)
{
  int idx = blockIdx.x * blockDim.x + threadIdx.x;
  if (idx >= B_ * NFM * HW) return;
  int hw = idx % HW;
  int f  = (idx / HW) % NFM;
  int b  = idx / (HW * NFM);
  int ch = fmidx[b * NFM + f];
  O[idx] = Ab[(b * C_ + ch) * HW + hw] + 0.1f * L2[idx];
}

// --------------------------------------------------------------------------
// Workspace layout (256B-aligned, total ~75.5 MB):
//   Ab_f32  @ 0          25,690,112 B   [b][c][56][56]
//   Pab_f16 @ 25690112   16,777,216 B   [c][b][64][64] zero-haloed
//   Amp_f16 @ 42467328   16,777,216 B   [c][b][64][64] masked
//   pA2t    @ 59244544    3,686,400 B   [b][60][60][32]
//   KcP     @ 62930944    6,553,600 B   4 x partial K_change
//   Kc_f32  @ 69484544    1,638,400 B
//   Kn2_f16 @ 71122944      819,200 B
//   Wg_f16  @ 71942144      819,200 B
//   L2_f32  @ 72761344    6,422,528 B
//   ssum    @ 79183872        8,192 B
//   winners @ 79192064        2,048 B
//   fmidx   @ 79194112        2,048 B
// --------------------------------------------------------------------------
extern "C" void kernel_launch(void* const* d_in, const int* in_sizes, int n_in,
                              void* d_out, int out_size, void* d_ws, size_t ws_size,
                              hipStream_t stream)
{
  (void)in_sizes; (void)n_in; (void)out_size; (void)ws_size;
  const float* A     = (const float*)d_in[0];
  const float* K     = (const float*)d_in[1];
  const float* noise = (const float*)d_in[2];

  char* ws = (char*)d_ws;
  float*    Ab      = (float*)(ws + 0);
  _Float16* Pab     = (_Float16*)(ws + 25690112);
  _Float16* Amp     = (_Float16*)(ws + 42467328);
  _Float16* pA2t    = (_Float16*)(ws + 59244544);
  float*    KcP     = (float*)(ws + 62930944);
  float*    Kc      = (float*)(ws + 69484544);
  _Float16* Kn2     = (_Float16*)(ws + 71122944);
  _Float16* Wg      = (_Float16*)(ws + 71942144);
  float*    L2      = (float*)(ws + 72761344);
  float*    ssum    = (float*)(ws + 79183872);
  int*      winners = (int*)(ws + 79192064);
  int*      fmidx   = (int*)(ws + 79194112);
  float*    O       = (float*)d_out;

  build_ab     <<<(B_ * C_ * HW + 255) / 256, 256, 0, stream>>>(A, noise, Ab);
  build_pab    <<<(C_ * CHS + 255) / 256, 256, 0, stream>>>(Ab, Pab);
  rowsum_kernel<<<B_ * C_, 256, 0, stream>>>(Ab, ssum);
  argmax_kernel<<<2, 256, 0, stream>>>(ssum, winners, fmidx);
  build_amp    <<<(C_ * CHS + 255) / 256, 256, 0, stream>>>(Pab, winners, Amp);
  einsum_wmma  <<<dim3(32, 25, NCHUNK), 32, 0, stream>>>(Amp, Pab, KcP);
  reduce_kc    <<<(KCN + 255) / 256, 256, 0, stream>>>(KcP, Kc);
  make_kn2     <<<(C_ * C_ + 255) / 256, 256, 0, stream>>>(K, Kc, Kn2);
  gather_w     <<<(B_ * NFM * NFM + 255) / 256, 256, 0, stream>>>(Kn2, fmidx, Wg);
  build_pa2t   <<<(B_ * PADW * PADW * NFM + 255) / 256, 256, 0, stream>>>(Ab, fmidx, pA2t);
  conv2_wmma   <<<dim3(196, B_), 32, 0, stream>>>(Wg, pA2t, L2);
  final_out    <<<(B_ * NFM * HW + 255) / 256, 256, 0, stream>>>(Ab, L2, fmidx, O);
}
